// GRUState_42090679501066
// MI455X (gfx1250) — compile-verified, matrix operand order checked by reading, not run on hardware
//
#include <hip/hip_runtime.h>

// ---------------------------------------------------------------------------
// GRU scan + projection for MI455X (gfx1250, wave32, WMMA bf16 path).
//
// Precision: bf16 operands, f32 WMMA accumulation (v_wmma_f32_16x16x32_bf16).
// All K dims padded to multiples of 32 AND all weight column dims padded per
// gate to 208 (13 exact 16-wide tiles) with zeros, so the WMMA inner loops
// contain no guards / EXEC branches at all: just aligned b128 loads + wmma.
// ---------------------------------------------------------------------------

#define B_DIM 1024
#define T_DIM 200
#define E_DIM 100
#define H_DIM 200
#define G3H   600          // 3*H (source weight layout)
#define GCPAD 208          // per-gate column dim padded to 13 tiles of 16
#define NITEMS 100000
#define EPAD  128          // E padded to multiple of 32
#define HPAD  224          // H padded to multiple of 32

typedef __attribute__((ext_vector_type(16))) __bf16 v16bf;
typedef __attribute__((ext_vector_type(8)))  __bf16 bf16x8;
typedef __attribute__((ext_vector_type(8)))  float  v8f;

__device__ __forceinline__ float sigmoidf_(float x) {
  return 1.0f / (1.0f + __expf(-x));
}

// Load a 16-element bf16 A/B fragment for one lane (unconditional, aligned).
// Lane mapping (CDNA5 16-bit 16x32 A layout): halves 0..7 -> K = k0+hi*8 ..+7,
// halves 8..15 -> K = k0+16+hi*8 ..+7.  (B assumed symmetric, N = lane&15.)
__device__ __forceinline__ v16bf load_frag(const __bf16* row, int k0, int hi) {
  bf16x8 c0 = *(const bf16x8*)(row + k0 + hi * 8);
  bf16x8 c1 = *(const bf16x8*)(row + k0 + 16 + hi * 8);
  v16bf a;
#pragma unroll
  for (int i = 0; i < 8; ++i) { a[i] = c0[i]; a[i + 8] = c1[i]; }
  return a;
}

// ---------------------------------------------------------------------------
// 1) state = bf16(embed_q[q] * embed_f[f]), stored time-major [T][B][EPAD].
// ---------------------------------------------------------------------------
__global__ void gather_embed_kernel(const int* __restrict__ q, const int* __restrict__ f,
                                    const float* __restrict__ eq, const float* __restrict__ ef,
                                    __bf16* __restrict__ X) {
  const int tb = blockIdx.x;            // tb = t*B + b
  const int t  = tb >> 10;              // /1024
  const int b  = tb & 1023;
  const int k  = threadIdx.x;           // 0..127
  const int qi = q[b * T_DIM + t];
  const int fi = f[b * T_DIM + t];
  float v = (k < E_DIM) ? eq[(size_t)qi * E_DIM + k] * ef[(size_t)fi * E_DIM + k] : 0.0f;
  X[(size_t)tb * EPAD + k] = (__bf16)v;
}

// ---------------------------------------------------------------------------
// 2a) Gate-aware LDS-tiled transpose + convert for Wi/Wh:
//     in[t][k][gate*200 + c] -> out[t][gate][GCPAD][Kpad], zero padded in both
//     the K tail (k >= R) and the per-gate column tail (c >= 200).
//     grid = (3*13, Kpad/16, T), block (16,16).
// ---------------------------------------------------------------------------
__global__ void transpose_gates_kernel(const float* __restrict__ in, __bf16* __restrict__ out,
                                       int R, int Kpad, long inSlice, long outSlice) {
  __shared__ float tile[16][17];
  const float* src = in + (long)blockIdx.z * inSlice;
  __bf16* dst = out + (long)blockIdx.z * outSlice;
  const int gate = blockIdx.x / 13;
  const int ct   = blockIdx.x % 13;     // column tile within gate
  const int c0 = ct * 16, k0 = blockIdx.y * 16;
  const int tx = threadIdx.x, ty = threadIdx.y;
  const int k = k0 + ty, c = c0 + tx;   // c = column within gate
  tile[ty][tx] = (k < R && c < H_DIM) ? src[(long)k * G3H + gate * H_DIM + c] : 0.0f;
  __syncthreads();
  const int oc = c0 + ty, ok = k0 + tx; // oc < GCPAD by construction
  dst[((long)gate * GCPAD + oc) * Kpad + ok] = (__bf16)tile[tx][ty];
}

// ---------------------------------------------------------------------------
// 2b) Generic transpose + convert for Wout: in[k][c] -> out[c][Kpad].
//     grid = (C/16, Kpad/16, 1), C multiple of 16.
// ---------------------------------------------------------------------------
__global__ void transpose_cvt_kernel(const float* __restrict__ in, __bf16* __restrict__ out,
                                     int R, int C, int Kpad) {
  __shared__ float tile[16][17];
  const int c0 = blockIdx.x * 16, k0 = blockIdx.y * 16;
  const int tx = threadIdx.x, ty = threadIdx.y;
  const int k = k0 + ty, c = c0 + tx;
  tile[ty][tx] = (k < R) ? in[(long)k * C + c] : 0.0f;
  __syncthreads();
  const int oc = c0 + ty, ok = k0 + tx;
  out[(long)oc * Kpad + ok] = (__bf16)tile[tx][ty];
}

// ---------------------------------------------------------------------------
// 3) h0 init: f32 [B][H] and bf16 [B][HPAD] (zero pad).
// ---------------------------------------------------------------------------
__global__ void init_h_kernel(const float* __restrict__ carry,
                              float* __restrict__ h32, __bf16* __restrict__ h16) {
  const int b = blockIdx.x;
  const int j = threadIdx.x;            // 0..223
  float v = (j < H_DIM) ? carry[j] : 0.0f;
  if (j < H_DIM) h32[b * H_DIM + j] = v;
  h16[b * HPAD + j] = (__bf16)v;
}

// ---------------------------------------------------------------------------
// 4) One GRU timestep.  Block = 128 threads (4 waves); wave -> 16x16 (b,j) tile.
//    grid = (B/64, HPAD/16) = (16, 14).  Tiles j0 <= 192 run branch-free WMMA
//    loops (weights zero-padded per gate to GCPAD); tile j0 = 208 only zeroes
//    the bf16 pad of h_out.
// ---------------------------------------------------------------------------
__global__ void gru_step_kernel(const __bf16* __restrict__ X,     // [T][B][EPAD]
                                const __bf16* __restrict__ WiT,   // [T][3][GCPAD][EPAD]
                                const __bf16* __restrict__ WhT,   // [T][3][GCPAD][HPAD]
                                const float*  __restrict__ bi,    // [T][600]
                                const float*  __restrict__ bhn,   // [T][200]
                                const float*  __restrict__ h_in32,
                                const __bf16* __restrict__ h_in16,
                                float*  __restrict__ h_out32,
                                __bf16* __restrict__ h_out16,
                                int t) {
  const int lane = threadIdx.x & 31;
  const int lo = lane & 15;             // N / M-low index
  const int hi = lane >> 4;             // K-chunk select (A/B), M-high (C/D)
  const int w  = threadIdx.x >> 5;
  const int b0 = blockIdx.x * 64 + w * 16;
  const int j0 = blockIdx.y * 16;

  if (j0 >= GCPAD) {                    // pad tile j0==208: zero bf16 pad, done
#pragma unroll
    for (int r = 0; r < 8; ++r)
      h_out16[(size_t)(b0 + r + 8 * hi) * HPAD + j0 + lo] = (__bf16)0.0f;
    return;
  }

  const int jcol = j0 + lo;             // < GCPAD always on this path
  const bool cv = (jcol < H_DIM);

  v8f acc_r = {}, acc_z = {}, acc_in = {}, acc_hn = {};

  // --- gi: x_t[16xE] @ Wi_t  (K padded to 128; gate cols padded to GCPAD)
  {
    const __bf16* arow = X + ((size_t)t * B_DIM + b0 + lo) * EPAD;
    const __bf16* wr = WiT + ((size_t)(t * 3 + 0) * GCPAD + jcol) * EPAD;
    const __bf16* wz = WiT + ((size_t)(t * 3 + 1) * GCPAD + jcol) * EPAD;
    const __bf16* wn = WiT + ((size_t)(t * 3 + 2) * GCPAD + jcol) * EPAD;
#pragma unroll
    for (int k0 = 0; k0 < EPAD; k0 += 32) {
      v16bf a  = load_frag(arow, k0, hi);
      v16bf br = load_frag(wr, k0, hi);
      v16bf bz = load_frag(wz, k0, hi);
      v16bf bn = load_frag(wn, k0, hi);
      acc_r  = __builtin_amdgcn_wmma_f32_16x16x32_bf16(false, a, false, br, (short)0, acc_r,  false, false);
      acc_z  = __builtin_amdgcn_wmma_f32_16x16x32_bf16(false, a, false, bz, (short)0, acc_z,  false, false);
      acc_in = __builtin_amdgcn_wmma_f32_16x16x32_bf16(false, a, false, bn, (short)0, acc_in, false, false);
    }
  }

  // --- gh: h[16xH] @ Wh_t  (K padded to 224; gate cols padded to GCPAD)
  {
    const __bf16* arow = h_in16 + (size_t)(b0 + lo) * HPAD;
    const __bf16* wr = WhT + ((size_t)(t * 3 + 0) * GCPAD + jcol) * HPAD;
    const __bf16* wz = WhT + ((size_t)(t * 3 + 1) * GCPAD + jcol) * HPAD;
    const __bf16* wn = WhT + ((size_t)(t * 3 + 2) * GCPAD + jcol) * HPAD;
#pragma unroll
    for (int k0 = 0; k0 < HPAD; k0 += 32) {
      v16bf a  = load_frag(arow, k0, hi);
      v16bf br = load_frag(wr, k0, hi);
      v16bf bz = load_frag(wz, k0, hi);
      v16bf bn = load_frag(wn, k0, hi);
      acc_r  = __builtin_amdgcn_wmma_f32_16x16x32_bf16(false, a, false, br, (short)0, acc_r,  false, false);
      acc_z  = __builtin_amdgcn_wmma_f32_16x16x32_bf16(false, a, false, bz, (short)0, acc_z,  false, false);
      acc_hn = __builtin_amdgcn_wmma_f32_16x16x32_bf16(false, a, false, bn, (short)0, acc_hn, false, false);
    }
  }

  // --- gates + state update.  C/D layout: M = r + 8*hi, N = lo.
  const float bir = cv ? bi[(size_t)t * G3H + 0 * H_DIM + jcol] : 0.0f;
  const float biz = cv ? bi[(size_t)t * G3H + 1 * H_DIM + jcol] : 0.0f;
  const float bin = cv ? bi[(size_t)t * G3H + 2 * H_DIM + jcol] : 0.0f;
  const float bhj = cv ? bhn[(size_t)t * H_DIM + jcol] : 0.0f;

#pragma unroll
  for (int r = 0; r < 8; ++r) {
    const int brow = b0 + r + 8 * hi;
    float rg = sigmoidf_(acc_r[r] + bir);
    float zg = sigmoidf_(acc_z[r] + biz);
    float ng = tanhf(acc_in[r] + bin + rg * (acc_hn[r] + bhj));
    float hold = cv ? h_in32[(size_t)brow * H_DIM + jcol] : 0.0f;
    float hnew = (1.0f - zg) * ng + zg * hold;
    if (cv) h_out32[(size_t)brow * H_DIM + jcol] = hnew;
    h_out16[(size_t)brow * HPAD + jcol] = cv ? (__bf16)hnew : (__bf16)0.0f;  // keeps pad zero
  }
}

// ---------------------------------------------------------------------------
// 5) out = hT @ Wout + bout.  64 x 6250 exact 16x16 tiles, K padded to 224.
//    Block = 256 threads (8 waves, 8 M-tiles); grid = (6250, 8).
// ---------------------------------------------------------------------------
__global__ void out_gemm_kernel(const __bf16* __restrict__ Hb,     // [B][HPAD]
                                const __bf16* __restrict__ WoutT,  // [NITEMS][HPAD]
                                const float*  __restrict__ bout,
                                float* __restrict__ out) {
  const int lane = threadIdx.x & 31;
  const int lo = lane & 15, hi = lane >> 4;
  const int w  = threadIdx.x >> 5;
  const int m_tile = blockIdx.y * 8 + w;      // 0..63
  const int n_tile = blockIdx.x;              // 0..6249
  const int b0 = m_tile * 16;
  const int c0 = n_tile * 16;

  const __bf16* arow = Hb + (size_t)(b0 + lo) * HPAD;
  const __bf16* brow = WoutT + (size_t)(c0 + lo) * HPAD;

  v8f acc = {};
#pragma unroll
  for (int k0 = 0; k0 < HPAD; k0 += 32) {
    v16bf a = load_frag(arow, k0, hi);
    v16bf b = load_frag(brow, k0, hi);
    acc = __builtin_amdgcn_wmma_f32_16x16x32_bf16(false, a, false, b, (short)0, acc, false, false);
  }

  const int col = c0 + lo;
  const float bb = bout[col];
#pragma unroll
  for (int r = 0; r < 8; ++r) {
    const int brow_i = b0 + r + 8 * hi;
    out[(size_t)brow_i * NITEMS + col] = acc[r] + bb;
  }
}

// ---------------------------------------------------------------------------
// Host: workspace layout + launch sequence (all on `stream`, graph-safe).
// ---------------------------------------------------------------------------
extern "C" void kernel_launch(void* const* d_in, const int* in_sizes, int n_in,
                              void* d_out, int out_size, void* d_ws, size_t ws_size,
                              hipStream_t stream) {
  const int*   q     = (const int*)  d_in[0];
  const int*   f     = (const int*)  d_in[1];
  const float* eq    = (const float*)d_in[2];
  const float* ef    = (const float*)d_in[3];
  const float* carry = (const float*)d_in[4];
  const float* Wi    = (const float*)d_in[5];
  const float* bi    = (const float*)d_in[6];
  const float* Wh    = (const float*)d_in[7];
  const float* bhn   = (const float*)d_in[8];
  const float* Wout  = (const float*)d_in[9];
  const float* bout  = (const float*)d_in[10];
  float* out = (float*)d_out;

  // Workspace layout (256B aligned blocks), total ~190 MB.
  char* ws = (char*)d_ws;
  size_t off = 0;
  auto alloc = [&](size_t bytes) { void* p = ws + off; off = (off + bytes + 255) & ~(size_t)255; return p; };
  __bf16* X     = (__bf16*)alloc((size_t)T_DIM * B_DIM * EPAD * 2);        // 52.4 MB
  __bf16* WiT   = (__bf16*)alloc((size_t)T_DIM * 3 * GCPAD * EPAD * 2);    // 31.9 MB
  __bf16* WhT   = (__bf16*)alloc((size_t)T_DIM * 3 * GCPAD * HPAD * 2);    // 55.9 MB
  __bf16* WoutT = (__bf16*)alloc((size_t)NITEMS * HPAD * 2);               // 44.8 MB
  float*  h32_0 = (float*) alloc((size_t)B_DIM * H_DIM * 4);
  float*  h32_1 = (float*) alloc((size_t)B_DIM * H_DIM * 4);
  __bf16* h16_0 = (__bf16*)alloc((size_t)B_DIM * HPAD * 2);
  __bf16* h16_1 = (__bf16*)alloc((size_t)B_DIM * HPAD * 2);
  float*  h32[2]  = {h32_0, h32_1};
  __bf16* h16[2]  = {h16_0, h16_1};

  // 1) embedding gather -> bf16 state, time-major, K-padded.
  gather_embed_kernel<<<T_DIM * B_DIM, EPAD, 0, stream>>>(q, f, eq, ef, X);

  // 2) weight transposes + bf16 convert (zero-padded K and gate columns).
  dim3 tb(16, 16);
  transpose_gates_kernel<<<dim3(3 * 13, EPAD / 16, T_DIM), tb, 0, stream>>>(
      Wi, WiT, E_DIM, EPAD, (long)E_DIM * G3H, (long)3 * GCPAD * EPAD);
  transpose_gates_kernel<<<dim3(3 * 13, HPAD / 16, T_DIM), tb, 0, stream>>>(
      Wh, WhT, H_DIM, HPAD, (long)H_DIM * G3H, (long)3 * GCPAD * HPAD);
  transpose_cvt_kernel<<<dim3(NITEMS / 16, HPAD / 16, 1), tb, 0, stream>>>(
      Wout, WoutT, H_DIM, NITEMS, HPAD);

  // 3) h0 init (both precisions; bf16 pad zeroed).
  init_h_kernel<<<B_DIM, HPAD, 0, stream>>>(carry, h32[0], h16[0]);

  // 4) sequential GRU scan: step t reads buf[t&1], writes buf[(t+1)&1].
  for (int t = 0; t < T_DIM; ++t) {
    gru_step_kernel<<<dim3(B_DIM / 64, HPAD / 16), 128, 0, stream>>>(
        X, WiT, WhT, bi, bhn,
        h32[t & 1], h16[t & 1], h32[(t + 1) & 1], h16[(t + 1) & 1], t);
  }

  // 5) projection: hT (in buffer 0 after 200 steps) @ Wout + bout.
  out_gemm_kernel<<<dim3(NITEMS / 16, 8), 256, 0, stream>>>(h16[0], WoutT, bout, out);
}
